// GAT_32676111188648
// MI455X (gfx1250) — matrix operand order repaired and checked by reading, not compiled, not run on hardware
//
#include <hip/hip_runtime.h>
#include <hip/hip_bf16.h>
#include <math.h>

typedef __attribute__((ext_vector_type(16))) _Float16 v16h;
typedef __attribute__((ext_vector_type(8)))  _Float16 v8h;
typedef __attribute__((ext_vector_type(8)))  float    v8f;

#define HID 128
#define NHEAD 4
#define CHAN 32
#define NEG_SLOPE 0.2f
#define BN_EPS 1e-5f

// ---------- helpers: order-preserving float<->uint encoding for atomicMax ----
__device__ __forceinline__ unsigned int enc_f32(float f) {
    unsigned int u = __float_as_uint(f);
    return (u & 0x80000000u) ? ~u : (u | 0x80000000u);
}
__device__ __forceinline__ float dec_f32(unsigned int u) {
    unsigned int b = (u & 0x80000000u) ? (u & 0x7fffffffu) : ~u;
    return __uint_as_float(b);
}

// ---------- generic fill -----------------------------------------------------
__global__ void fill_u32_kernel(unsigned int* __restrict__ p, unsigned int v, long long n) {
    long long i = (long long)blockIdx.x * blockDim.x + threadIdx.x;
    if (i < n) p[i] = v;
}

// ---------- WMMA GEMM: H[N,128] = X[N,128] @ W[128,128]  (f16 in, f32 acc) ---
// Block = 256 threads (8 waves). Block computes a 128-row x 128-col tile.
// Wave w owns rows [row0 + w*16, +16) and loops over all 8 column tiles,
// reusing its A fragment across the 8 WMMAs of each K-step (32 WMMA / wave).
// W is staged into LDS pre-swizzled into the exact CDNA5 B-fragment layout so
// each B fragment is one aligned 32B vector load (2x ds_load_b128).
__global__ __launch_bounds__(256) void gemm_wmma_kernel(
    const float* __restrict__ X, const float* __restrict__ W,
    float* __restrict__ Hout, int n_rows)
{
    __shared__ _Float16 xs[128 * HID];       // 32 KB : X strip, row-major
    __shared__ _Float16 wsf[128 * HID];      // 32 KB : W in B-fragment layout

    const int tid  = threadIdx.x;
    const int row0 = blockIdx.x * 128;

    // ---- stage W swizzled: fragment-row fr = (nt*4 + ks)*32 + lane ----------
    // halves i=0..15 of fr hold W[k][n] with k = ks*32 + (lane>>4)*8 + (i&7) + (i>>3)*16,
    //                                      n = nt*16 + (lane&15)
    for (int fr = tid; fr < 1024; fr += 256) {
        int lf  = fr & 31;
        int ks  = (fr >> 5) & 3;
        int nt  = fr >> 7;
        int gf  = lf >> 4;
        int n   = nt * 16 + (lf & 15);
        int kb  = ks * 32 + gf * 8;
        v16h tmp;
        #pragma unroll
        for (int i = 0; i < 16; ++i)
            tmp[i] = (_Float16)W[(kb + (i & 7) + ((i >> 3) << 4)) * HID + n];
        *(v16h*)&wsf[fr * 16] = tmp;
    }

    // ---- stage X strip (float4 global loads, clamp OOB rows) ----------------
    {
        const float4* X4 = (const float4*)X;
        for (int idx = tid; idx < 128 * 32; idx += 256) {
            int r  = idx >> 5;
            int c4 = idx & 31;
            int row = row0 + r;
            long long rr = (row < n_rows) ? row : (n_rows - 1);
            float4 v = X4[rr * 32 + c4];
            _Float16* dst = &xs[r * HID + c4 * 4];
            dst[0] = (_Float16)v.x; dst[1] = (_Float16)v.y;
            dst[2] = (_Float16)v.z; dst[3] = (_Float16)v.w;
        }
    }
    __syncthreads();

    const int wave = tid >> 5;
    const int lane = tid & 31;
    const int g    = lane >> 4;
    const int ml   = lane & 15;
    const int mrow = wave * 16 + ml;         // A-row within strip for this lane

    v8f acc[8];
    #pragma unroll
    for (int nt = 0; nt < 8; ++nt) acc[nt] = (v8f){};

    #pragma unroll
    for (int ks = 0; ks < 4; ++ks) {         // K = 128 in 4 steps of 32
        const int kbase = ks * 32;
        // A fragment: halves 0..7 -> K = kbase + g*8 + i (contiguous)
        //             halves 8..15 -> K = kbase + 16 + g*8 + i (contiguous)
        v8h alo = *(const v8h*)&xs[mrow * HID + kbase + g * 8];
        v8h ahi = *(const v8h*)&xs[mrow * HID + kbase + 16 + g * 8];
        v16h a = __builtin_shufflevector(alo, ahi,
                 0,1,2,3,4,5,6,7,8,9,10,11,12,13,14,15);
        #pragma unroll
        for (int nt = 0; nt < 8; ++nt) {
            v16h b = *(const v16h*)&wsf[(((nt << 2) | ks) * 32 + lane) * 16];
            acc[nt] = __builtin_amdgcn_wmma_f32_16x16x32_f16(
                false, a, false, b, (short)0, acc[nt], false, false);
        }
    }

    // ---- store: C/D layout: VGPR r -> row (r + 8*g), col = nt*16 + ml -------
    #pragma unroll
    for (int nt = 0; nt < 8; ++nt) {
        #pragma unroll
        for (int r = 0; r < 8; ++r) {
            int row = row0 + wave * 16 + r + 8 * g;
            if (row < n_rows)
                Hout[(long long)row * HID + nt * 16 + ml] = acc[nt][r];
        }
    }
}

// ---------- attention logits: e_src[n,h] = <h[n,h,:], att_src[h,:]> ----------
__global__ void att_logits_kernel(const float* __restrict__ H,
                                  const float* __restrict__ a_src,
                                  const float* __restrict__ a_dst,
                                  float* __restrict__ e_src,
                                  float* __restrict__ e_dst, int n_rows)
{
    int tid = blockIdx.x * blockDim.x + threadIdx.x;
    if (tid >= n_rows * NHEAD) return;
    int n = tid >> 2, h0 = tid & 3;
    const float* hp = H + (long long)n * HID + h0 * CHAN;
    const float* as = a_src + h0 * CHAN;
    const float* ad = a_dst + h0 * CHAN;
    float ssum = 0.f, dsum = 0.f;
    #pragma unroll
    for (int c = 0; c < CHAN; ++c) {
        float v = hp[c];
        ssum += v * as[c];
        dsum += v * ad[c];
    }
    e_src[tid] = ssum;
    e_dst[tid] = dsum;
}

// ---------- edge pass 1: segment max of leaky_relu(es[src]+ed[dst]) ----------
__global__ void edge_max_kernel(const int* __restrict__ ei,
                                const float* __restrict__ es,
                                const float* __restrict__ ed,
                                unsigned int* __restrict__ m,
                                int E_, int EP)
{
    int tid = blockIdx.x * blockDim.x + threadIdx.x;
    if (tid >= EP * NHEAD) return;
    int e = tid >> 2, h0 = tid & 3;
    int s, d;
    if (e < E_) { s = ei[e]; d = ei[E_ + e]; } else { s = d = e - E_; }
    float v = es[s * NHEAD + h0] + ed[d * NHEAD + h0];
    v = (v > 0.f) ? v : NEG_SLOPE * v;
    atomicMax(&m[d * NHEAD + h0], enc_f32(v));
}

// ---------- edge pass 2: ex = exp(e - m[dst]); den[dst] += ex ----------------
__global__ void edge_expsum_kernel(const int* __restrict__ ei,
                                   const float* __restrict__ es,
                                   const float* __restrict__ ed,
                                   const unsigned int* __restrict__ m,
                                   float* __restrict__ exbuf,
                                   float* __restrict__ den,
                                   int E_, int EP)
{
    int tid = blockIdx.x * blockDim.x + threadIdx.x;
    if (tid >= EP * NHEAD) return;
    int e = tid >> 2, h0 = tid & 3;
    int s, d;
    if (e < E_) { s = ei[e]; d = ei[E_ + e]; } else { s = d = e - E_; }
    float v = es[s * NHEAD + h0] + ed[d * NHEAD + h0];
    v = (v > 0.f) ? v : NEG_SLOPE * v;
    float ex = __expf(v - dec_f32(m[d * NHEAD + h0]));
    exbuf[tid] = ex;
    atomicAdd(&den[d * NHEAD + h0], ex);
}

// ---------- edge pass 3: out[dst,c] += h[src,c] * alpha ----------------------
__global__ void edge_aggregate_kernel(const int* __restrict__ ei,
                                      const float* __restrict__ H,
                                      const float* __restrict__ exbuf,
                                      const float* __restrict__ den,
                                      float* __restrict__ out,
                                      int E_, int EP)
{
    long long tid = (long long)blockIdx.x * blockDim.x + threadIdx.x;
    if (tid >= (long long)EP * HID) return;
    int e = (int)(tid >> 7);
    int c = (int)(tid & (HID - 1));
    int h0 = c >> 5;
    int s, d;
    if (e < E_) { s = ei[e]; d = ei[E_ + e]; } else { s = d = e - E_; }
    float alpha = exbuf[e * NHEAD + h0] / (den[d * NHEAD + h0] + 1e-16f);
    atomicAdd(&out[(long long)d * HID + c], H[(long long)s * HID + c] * alpha);
}

// ---------- BatchNorm stats (per-channel sum / sumsq over nodes) -------------
#define BN_ROWS 256
__global__ __launch_bounds__(128) void bn_stats_kernel(
    const float* __restrict__ H, const float* __restrict__ bias,
    float* __restrict__ sums, float* __restrict__ sqs, int n_rows)
{
    int c = threadIdx.x;
    float b = bias[c];
    long long r0 = (long long)blockIdx.x * BN_ROWS;
    float s = 0.f, q = 0.f;
    for (int i = 0; i < BN_ROWS; ++i) {
        long long r = r0 + i;
        if (r < n_rows) {
            if (i + 8 < BN_ROWS && r + 8 < n_rows)
                __builtin_prefetch(&H[(r + 8) * HID + c], 0, 0);  // global_prefetch_b8
            float v = H[r * HID + c] + b;
            s += v;
            q += v * v;
        }
    }
    atomicAdd(&sums[c], s);
    atomicAdd(&sqs[c], q);
}

// ---------- BatchNorm apply + ELU (in place) ---------------------------------
__global__ void bn_apply_elu_kernel(float* __restrict__ H,
                                    const float* __restrict__ bias,
                                    const float* __restrict__ sums,
                                    const float* __restrict__ sqs,
                                    const float* __restrict__ gamma,
                                    const float* __restrict__ beta,
                                    int n_rows)
{
    long long tid = (long long)blockIdx.x * blockDim.x + threadIdx.x;
    if (tid >= (long long)n_rows * HID) return;
    int c = (int)(tid & (HID - 1));
    float invn = 1.0f / (float)n_rows;
    float mean = sums[c] * invn;
    float var  = sqs[c] * invn - mean * mean;
    float v = (H[tid] + bias[c] - mean) * rsqrtf(var + BN_EPS) * gamma[c] + beta[c];
    H[tid] = (v > 0.f) ? v : expm1f(v);
}

// ---------- global mean pool: accumulate sums + counts -----------------------
__global__ void pool_kernel(const float* __restrict__ X,
                            const int* __restrict__ batch,
                            float* __restrict__ pooled,
                            float* __restrict__ cnt, int n_rows)
{
    long long tid = (long long)blockIdx.x * blockDim.x + threadIdx.x;
    if (tid >= (long long)n_rows * HID) return;
    int node = (int)(tid >> 7);
    int c    = (int)(tid & (HID - 1));
    int b = batch[node];
    atomicAdd(&pooled[(long long)b * HID + c], X[tid]);
    if (c == 0) atomicAdd(&cnt[b], 1.0f);
}

// ---------- fused MLP head: one wave32 per group -----------------------------
__global__ __launch_bounds__(32) void mlp_kernel(const float* __restrict__ pooled,
                                                 const float* __restrict__ cnt,
                                                 const float* __restrict__ w1,
                                                 const float* __restrict__ b1,
                                                 const float* __restrict__ w2,
                                                 const float* __restrict__ b2,
                                                 float* __restrict__ out)
{
    __shared__ float p[HID];
    int g = blockIdx.x;
    int lane = threadIdx.x;
    float invc = 1.0f / fmaxf(cnt[g], 1.0f);
    for (int c = lane; c < HID; c += 32) p[c] = pooled[(long long)g * HID + c] * invc;
    __syncthreads();
    float acc = 0.f;
    #pragma unroll
    for (int jj = 0; jj < 2; ++jj) {
        int j = lane + jj * 32;
        float dot = b1[j];
        for (int c = 0; c < HID; ++c) dot += p[c] * w1[c * 64 + j];
        dot = fmaxf(dot, 0.f);
        acc += dot * w2[j];
    }
    #pragma unroll
    for (int off = 16; off > 0; off >>= 1) acc += __shfl_down(acc, off, 32);
    if (lane == 0) out[g] = acc + b2[0];
}

// =============================================================================
extern "C" void kernel_launch(void* const* d_in, const int* in_sizes, int n_in,
                              void* d_out, int out_size, void* d_ws, size_t ws_size,
                              hipStream_t stream)
{
    const float* x_in   = (const float*)d_in[0];
    const int*   ei     = (const int*)d_in[1];
    const int*   batch  = (const int*)d_in[2];
    const float* Ws     = (const float*)d_in[3];
    const float* attS   = (const float*)d_in[4];
    const float* attD   = (const float*)d_in[5];
    const float* biases = (const float*)d_in[6];
    const float* gammas = (const float*)d_in[7];
    const float* betas  = (const float*)d_in[8];
    const float* lin1w  = (const float*)d_in[9];
    const float* lin1b  = (const float*)d_in[10];
    const float* lin2w  = (const float*)d_in[11];
    const float* lin2b  = (const float*)d_in[12];
    float* out = (float*)d_out;

    const int N  = in_sizes[2];          // 50000
    const int E  = in_sizes[1] / 2;      // 800000
    const int EP = E + N;                // edges incl. self loops
    const int G  = out_size;             // 1024
    const int L  = 3;

    // ---- workspace layout (256B aligned) ----
    char* base = (char*)d_ws;
    size_t off = 0;
    auto alloc = [&](size_t bytes) -> char* {
        char* p = base + off;
        off = (off + bytes + 255) & ~(size_t)255;
        return p;
    };
    float*        hbuf   = (float*)alloc((size_t)N * HID * 4);   // h = xW
    float*        obuf   = (float*)alloc((size_t)N * HID * 4);   // aggregated / next x
    float*        e_src  = (float*)alloc((size_t)N * NHEAD * 4);
    float*        e_dst  = (float*)alloc((size_t)N * NHEAD * 4);
    unsigned int* mbuf   = (unsigned int*)alloc((size_t)N * NHEAD * 4);
    float*        den    = (float*)alloc((size_t)N * NHEAD * 4);
    float*        exbuf  = (float*)alloc((size_t)EP * NHEAD * 4);
    float*        sums   = (float*)alloc(HID * 4);
    float*        sqs    = (float*)alloc(HID * 4);
    float*        pooled = (float*)alloc((size_t)G * HID * 4);
    float*        cnt    = (float*)alloc((size_t)G * 4);
    (void)ws_size;

    const unsigned int ENC_NEG_INF = 0x007FFFFFu;   // enc_f32(-inf)

    auto cdiv = [](long long a, long long b) { return (int)((a + b - 1) / b); };

    for (int l = 0; l < L; ++l) {
        const float* xl = (l == 0) ? x_in : obuf;
        const float* Wl = Ws + (size_t)l * HID * HID;

        // 1) h = x @ W   (WMMA, 128-row blocks)
        gemm_wmma_kernel<<<cdiv(N, 128), 256, 0, stream>>>(xl, Wl, hbuf, N);

        // 2) attention logits
        att_logits_kernel<<<cdiv((long long)N * NHEAD, 256), 256, 0, stream>>>(
            hbuf, attS + (size_t)l * HID, attD + (size_t)l * HID, e_src, e_dst, N);

        // 3) init accumulators (x already consumed into hbuf)
        fill_u32_kernel<<<cdiv((long long)N * NHEAD, 256), 256, 0, stream>>>(
            mbuf, ENC_NEG_INF, (long long)N * NHEAD);
        fill_u32_kernel<<<cdiv((long long)N * NHEAD, 256), 256, 0, stream>>>(
            (unsigned int*)den, 0u, (long long)N * NHEAD);
        fill_u32_kernel<<<cdiv((long long)N * HID, 256), 256, 0, stream>>>(
            (unsigned int*)obuf, 0u, (long long)N * HID);

        // 4-6) segment softmax + weighted aggregation over edges
        edge_max_kernel<<<cdiv((long long)EP * NHEAD, 256), 256, 0, stream>>>(
            ei, e_src, e_dst, mbuf, E, EP);
        edge_expsum_kernel<<<cdiv((long long)EP * NHEAD, 256), 256, 0, stream>>>(
            ei, e_src, e_dst, mbuf, exbuf, den, E, EP);
        edge_aggregate_kernel<<<cdiv((long long)EP * HID, 256), 256, 0, stream>>>(
            ei, hbuf, exbuf, den, obuf, E, EP);

        // 7) BatchNorm (+bias, batch stats) + ELU, in place -> next layer input
        fill_u32_kernel<<<1, 256, 0, stream>>>((unsigned int*)sums, 0u, HID);
        fill_u32_kernel<<<1, 256, 0, stream>>>((unsigned int*)sqs, 0u, HID);
        bn_stats_kernel<<<cdiv(N, BN_ROWS), 128, 0, stream>>>(
            obuf, biases + (size_t)l * HID, sums, sqs, N);
        bn_apply_elu_kernel<<<cdiv((long long)N * HID, 256), 256, 0, stream>>>(
            obuf, biases + (size_t)l * HID, sums, sqs,
            gammas + (size_t)l * HID, betas + (size_t)l * HID, N);
    }

    // ---- global mean pool ----
    fill_u32_kernel<<<cdiv((long long)G * HID, 256), 256, 0, stream>>>(
        (unsigned int*)pooled, 0u, (long long)G * HID);
    fill_u32_kernel<<<cdiv(G, 256), 256, 0, stream>>>((unsigned int*)cnt, 0u, G);
    pool_kernel<<<cdiv((long long)N * HID, 256), 256, 0, stream>>>(
        obuf, batch, pooled, cnt, N);

    // ---- MLP head ----
    mlp_kernel<<<G, 32, 0, stream>>>(pooled, cnt, lin1w, lin1b, lin2w, lin2b, out);
}
